// GNNL_56487409877429
// MI455X (gfx1250) — matrix operand order repaired and checked by reading, not compiled, not run on hardware
//
#include <hip/hip_runtime.h>
#include <hip/hip_bf16.h>
#include <stdint.h>

typedef _Float16 v8h  __attribute__((ext_vector_type(8)));
typedef _Float16 v16h __attribute__((ext_vector_type(16)));
typedef float    v8f  __attribute__((ext_vector_type(8)));

#define N_NODES 50000
#define N_GRAPHS 64
#define HDIM 256

// ---------------- utility kernels ----------------

__global__ void k_fill(float* __restrict__ p, int n, float v) {
    int i = blockIdx.x * blockDim.x + threadIdx.x;
    if (i < n) p[i] = v;
}

__global__ void k_deg(const int* __restrict__ dst, int E, float* __restrict__ deg) {
    int i = blockIdx.x * blockDim.x + threadIdx.x;
    if (i < E) {
        __hip_atomic_fetch_add(&deg[dst[i]], 1.0f, __ATOMIC_RELAXED, __HIP_MEMORY_SCOPE_AGENT);
    }
}

__global__ void k_rsqrt_inplace(float* __restrict__ p, int n) {
    int i = blockIdx.x * blockDim.x + threadIdx.x;
    if (i < n) {
        float d = p[i];
        p[i] = (d > 0.0f) ? rsqrtf(d) : 0.0f;
    }
}

__global__ void k_f32_to_f16(const float* __restrict__ in, _Float16* __restrict__ out, int n) {
    int i = blockIdx.x * blockDim.x + threadIdx.x;
    if (i < n) out[i] = (_Float16)in[i];
}

// w: [K, 256] fp32 row-major -> wt: [256, K] f16 row-major (transposed)
__global__ void k_wtrans(const float* __restrict__ w, _Float16* __restrict__ wt, int K) {
    int i = blockIdx.x * blockDim.x + threadIdx.x;
    if (i < K * HDIM) {
        int k = i / HDIM, n = i % HDIM;
        wt[(size_t)n * K + k] = (_Float16)w[i];
    }
}

// ---------------- WMMA GEMM:  C[M,256] = A[M,K](f16) * Bt[256,K](f16, pre-transposed) ----------------
// block = 256 threads = 8 waves; wave w computes rows [ (bx*8+w)*16 , +16 ) x cols [by*64, +64)
__global__ void __launch_bounds__(256)
k_gemm_f16(const _Float16* __restrict__ A, const _Float16* __restrict__ Bt,
           float* __restrict__ C, int M, int K) {
    const int wave = threadIdx.x >> 5;
    const int lane = threadIdx.x & 31;
    const int m0 = (blockIdx.x * 8 + wave) * 16;
    const int n0 = blockIdx.y * 64;

    int row = m0 + (lane & 15);
    if (row >= M) row = M - 1;                  // clamp: keep EXEC all-ones for WMMA
    const int hi = (lane >= 16) ? 1 : 0;

    // A fragment base (contiguous K layout): halves 0..15 -> K + (hi?16:0) + 0..15
    const _Float16* arow = A + (size_t)row * K + (hi ? 16 : 0);
    // B fragment base: lanes 0..15 hold K=kb..kb+15, lanes 16..31 hold K=kb+16..kb+31
    const int ncol = n0 + (lane & 15);
    const _Float16* bp0 = Bt + (size_t)(ncol +  0) * K + (hi ? 16 : 0);
    const _Float16* bp1 = Bt + (size_t)(ncol + 16) * K + (hi ? 16 : 0);
    const _Float16* bp2 = Bt + (size_t)(ncol + 32) * K + (hi ? 16 : 0);
    const _Float16* bp3 = Bt + (size_t)(ncol + 48) * K + (hi ? 16 : 0);

    v8f acc0 = {}, acc1 = {}, acc2 = {}, acc3 = {};

    for (int kb = 0; kb < K; kb += 32) {
        v8h alo = *(const v8h*)(arow + kb);
        v8h ahi = *(const v8h*)(arow + kb + 8);
        v16h a;
#pragma unroll
        for (int i = 0; i < 8; ++i) { a[i] = alo[i]; a[i + 8] = ahi[i]; }

        v16h b0 = *(const v16h*)(bp0 + kb);
        v16h b1 = *(const v16h*)(bp1 + kb);
        v16h b2 = *(const v16h*)(bp2 + kb);
        v16h b3 = *(const v16h*)(bp3 + kb);

        acc0 = __builtin_amdgcn_wmma_f32_16x16x32_f16(false, a, false, b0, (short)0, acc0, false, false);
        acc1 = __builtin_amdgcn_wmma_f32_16x16x32_f16(false, a, false, b1, (short)0, acc1, false, false);
        acc2 = __builtin_amdgcn_wmma_f32_16x16x32_f16(false, a, false, b2, (short)0, acc2, false, false);
        acc3 = __builtin_amdgcn_wmma_f32_16x16x32_f16(false, a, false, b3, (short)0, acc3, false, false);
    }

    // C/D layout: lane 0-15 -> N = lane, VGPR r -> M = r ; lane 16-31 -> M = r + 8
    const int mofs = hi ? 8 : 0;
    const int cc = n0 + (lane & 15);
#pragma unroll
    for (int r = 0; r < 8; ++r) {
        int rr = m0 + r + mofs;
        if (rr < M) {
            float* crow = C + (size_t)rr * HDIM + cc;
            crow[0]  = acc0[r];
            crow[16] = acc1[r];
            crow[32] = acc2[r];
            crow[48] = acc3[r];
        }
    }
}

// ---------------- edge scatter:  agg[dst,:] += xw[src,:] * dinv[src]*dinv[dst]  ----------------
// one wave per (edge or self-loop); lane handles 8 consecutive features
__global__ void k_scatter(const int* __restrict__ src, const int* __restrict__ dst, int E, int Nn,
                          const float* __restrict__ dinv, const float* __restrict__ xw,
                          float* __restrict__ agg) {
    const int lane = threadIdx.x & 31;
    const int wave = (blockIdx.x * blockDim.x + threadIdx.x) >> 5;
    const int nwaves = (gridDim.x * blockDim.x) >> 5;
    const int T = E + Nn;
    for (int e = wave; e < T; e += nwaves) {
        int s, d;
        if (e < E) { s = src[e]; d = dst[e]; } else { s = e - E; d = s; }
        float norm = dinv[s] * dinv[d];
        const float4* ps = (const float4*)(xw + (size_t)s * HDIM) + lane * 2;
        float* pd = agg + (size_t)d * HDIM + lane * 8;
        float4 v0 = ps[0];
        float4 v1 = ps[1];
        __hip_atomic_fetch_add(pd + 0, v0.x * norm, __ATOMIC_RELAXED, __HIP_MEMORY_SCOPE_AGENT);
        __hip_atomic_fetch_add(pd + 1, v0.y * norm, __ATOMIC_RELAXED, __HIP_MEMORY_SCOPE_AGENT);
        __hip_atomic_fetch_add(pd + 2, v0.z * norm, __ATOMIC_RELAXED, __HIP_MEMORY_SCOPE_AGENT);
        __hip_atomic_fetch_add(pd + 3, v0.w * norm, __ATOMIC_RELAXED, __HIP_MEMORY_SCOPE_AGENT);
        __hip_atomic_fetch_add(pd + 4, v1.x * norm, __ATOMIC_RELAXED, __HIP_MEMORY_SCOPE_AGENT);
        __hip_atomic_fetch_add(pd + 5, v1.y * norm, __ATOMIC_RELAXED, __HIP_MEMORY_SCOPE_AGENT);
        __hip_atomic_fetch_add(pd + 6, v1.z * norm, __ATOMIC_RELAXED, __HIP_MEMORY_SCOPE_AGENT);
        __hip_atomic_fetch_add(pd + 7, v1.w * norm, __ATOMIC_RELAXED, __HIP_MEMORY_SCOPE_AGENT);
    }
}

// out(f16) = relu(agg + bias[f])   (bias added AFTER aggregation, per reference)
__global__ void k_finalize_relu_f16(const float* __restrict__ agg, const float* __restrict__ bias,
                                    _Float16* __restrict__ out, int n) {
    int i = blockIdx.x * blockDim.x + threadIdx.x;
    if (i < n) {
        float v = agg[i] + bias[i & (HDIM - 1)];
        out[i] = (_Float16)(v > 0.0f ? v : 0.0f);
    }
}

// out(f32) = agg + bias[f]   (last conv layer, no relu)
__global__ void k_finalize_f32(const float* __restrict__ agg, const float* __restrict__ bias,
                               float* __restrict__ out, int n) {
    int i = blockIdx.x * blockDim.x + threadIdx.x;
    if (i < n) out[i] = agg[i] + bias[i & (HDIM - 1)];
}

// ---------------- pooling + MLP ----------------
__global__ void k_pool(const float* __restrict__ h, const int* __restrict__ batch,
                       float* __restrict__ pooled, int Nn) {
    const int lane = threadIdx.x & 31;
    const int wave = (blockIdx.x * blockDim.x + threadIdx.x) >> 5;
    const int nwaves = (gridDim.x * blockDim.x) >> 5;
    for (int node = wave; node < Nn; node += nwaves) {
        int g = batch[node];
        const float4* ps = (const float4*)(h + (size_t)node * HDIM) + lane * 2;
        float* pd = pooled + (size_t)g * HDIM + lane * 8;
        float4 v0 = ps[0];
        float4 v1 = ps[1];
        __hip_atomic_fetch_add(pd + 0, v0.x, __ATOMIC_RELAXED, __HIP_MEMORY_SCOPE_AGENT);
        __hip_atomic_fetch_add(pd + 1, v0.y, __ATOMIC_RELAXED, __HIP_MEMORY_SCOPE_AGENT);
        __hip_atomic_fetch_add(pd + 2, v0.z, __ATOMIC_RELAXED, __HIP_MEMORY_SCOPE_AGENT);
        __hip_atomic_fetch_add(pd + 3, v0.w, __ATOMIC_RELAXED, __HIP_MEMORY_SCOPE_AGENT);
        __hip_atomic_fetch_add(pd + 4, v1.x, __ATOMIC_RELAXED, __HIP_MEMORY_SCOPE_AGENT);
        __hip_atomic_fetch_add(pd + 5, v1.y, __ATOMIC_RELAXED, __HIP_MEMORY_SCOPE_AGENT);
        __hip_atomic_fetch_add(pd + 6, v1.z, __ATOMIC_RELAXED, __HIP_MEMORY_SCOPE_AGENT);
        __hip_atomic_fetch_add(pd + 7, v1.w, __ATOMIC_RELAXED, __HIP_MEMORY_SCOPE_AGENT);
    }
}

__global__ void k_count(const int* __restrict__ batch, float* __restrict__ cnt, int Nn) {
    int i = blockIdx.x * blockDim.x + threadIdx.x;
    if (i < Nn) {
        __hip_atomic_fetch_add(&cnt[batch[i]], 1.0f, __ATOMIC_RELAXED, __HIP_MEMORY_SCOPE_AGENT);
    }
}

__global__ void k_div(float* __restrict__ pooled, const float* __restrict__ cnt, int n) {
    int i = blockIdx.x * blockDim.x + threadIdx.x;
    if (i < n) pooled[i] /= fmaxf(cnt[i / HDIM], 1.0f);
}

// out[G,N] = in[G,K] @ w[K,N] + b[N]
__global__ void k_mlp(const float* __restrict__ in, const float* __restrict__ w,
                      const float* __restrict__ b, float* __restrict__ out,
                      int G, int K, int N) {
    int idx = blockIdx.x * blockDim.x + threadIdx.x;
    if (idx >= G * N) return;
    int g = idx / N, j = idx % N;
    float s = b[j];
    const float* ir = in + (size_t)g * K;
    for (int k = 0; k < K; ++k) s += ir[k] * w[(size_t)k * N + j];
    out[idx] = s;
}

// ---------------- host launch ----------------

extern "C" void kernel_launch(void* const* d_in, const int* in_sizes, int n_in,
                              void* d_out, int out_size, void* d_ws, size_t ws_size,
                              hipStream_t stream) {
    const float* x      = (const float*)d_in[0];
    const int*   eidx   = (const int*)d_in[2];   // [2, E]
    const int*   batch  = (const int*)d_in[3];
    const float* w1 = (const float*)d_in[4];  const float* b1 = (const float*)d_in[5];
    const float* w2 = (const float*)d_in[6];  const float* b2 = (const float*)d_in[7];
    const float* w3 = (const float*)d_in[8];  const float* b3 = (const float*)d_in[9];
    const float* l1w = (const float*)d_in[10]; const float* l1b = (const float*)d_in[11];
    const float* l2w = (const float*)d_in[12]; const float* l2b = (const float*)d_in[13];

    const int E  = in_sizes[2] / 2;
    const int Nn = N_NODES;
    const int* src = eidx;
    const int* dst = eidx + E;

    // carve workspace
    char* wp = (char*)d_ws;
    auto carve = [&](size_t bytes) -> void* {
        void* r = (void*)wp;
        wp += (bytes + 511) & ~(size_t)511;
        return r;
    };
    float*     dinv   = (float*)    carve((size_t)Nn * 4);
    _Float16*  xh     = (_Float16*) carve((size_t)Nn * HDIM * 2);
    _Float16*  wt     = (_Float16*) carve((size_t)HDIM * HDIM * 2);
    float*     xw     = (float*)    carve((size_t)Nn * HDIM * 4);
    float*     agg    = (float*)    carve((size_t)Nn * HDIM * 4);
    float*     pooled = (float*)    carve((size_t)N_GRAPHS * HDIM * 4);
    float*     cnt    = (float*)    carve((size_t)N_GRAPHS * 4);
    float*     m1     = (float*)    carve((size_t)N_GRAPHS * 128 * 4);

    const int nh = Nn * HDIM;

    // degrees (self-loop contributes 1 -> init to 1), then dinv = rsqrt(deg)
    k_fill<<<(Nn + 255) / 256, 256, 0, stream>>>(dinv, Nn, 1.0f);
    k_deg<<<(E + 255) / 256, 256, 0, stream>>>(dst, E, dinv);
    k_rsqrt_inplace<<<(Nn + 255) / 256, 256, 0, stream>>>(dinv, Nn);

    // x -> f16
    k_f32_to_f16<<<(Nn * 128 + 255) / 256, 256, 0, stream>>>(x, xh, Nn * 128);

    dim3 gemmGrid((Nn + 127) / 128, HDIM / 64);

    // ---- layer 1: K=128, relu ----
    k_wtrans<<<(128 * HDIM + 255) / 256, 256, 0, stream>>>(w1, wt, 128);
    k_gemm_f16<<<gemmGrid, 256, 0, stream>>>(xh, wt, xw, Nn, 128);
    k_fill<<<(nh + 255) / 256, 256, 0, stream>>>(agg, nh, 0.0f);
    k_scatter<<<4096, 256, 0, stream>>>(src, dst, E, Nn, dinv, xw, agg);
    k_finalize_relu_f16<<<(nh + 255) / 256, 256, 0, stream>>>(agg, b1, xh, nh);

    // ---- layer 2: K=256, relu ----
    k_wtrans<<<(HDIM * HDIM + 255) / 256, 256, 0, stream>>>(w2, wt, HDIM);
    k_gemm_f16<<<gemmGrid, 256, 0, stream>>>(xh, wt, xw, Nn, HDIM);
    k_fill<<<(nh + 255) / 256, 256, 0, stream>>>(agg, nh, 0.0f);
    k_scatter<<<4096, 256, 0, stream>>>(src, dst, E, Nn, dinv, xw, agg);
    k_finalize_relu_f16<<<(nh + 255) / 256, 256, 0, stream>>>(agg, b2, xh, nh);

    // ---- layer 3: K=256, no relu, fp32 result into xw ----
    k_wtrans<<<(HDIM * HDIM + 255) / 256, 256, 0, stream>>>(w3, wt, HDIM);
    k_gemm_f16<<<gemmGrid, 256, 0, stream>>>(xh, wt, xw, Nn, HDIM);
    k_fill<<<(nh + 255) / 256, 256, 0, stream>>>(agg, nh, 0.0f);
    k_scatter<<<4096, 256, 0, stream>>>(src, dst, E, Nn, dinv, xw, agg);
    k_finalize_f32<<<(nh + 255) / 256, 256, 0, stream>>>(agg, b3, xw, nh);

    // ---- global mean pool ----
    k_fill<<<(N_GRAPHS * HDIM + 255) / 256, 256, 0, stream>>>(pooled, N_GRAPHS * HDIM, 0.0f);
    k_fill<<<1, 64, 0, stream>>>(cnt, N_GRAPHS, 0.0f);
    k_pool<<<1024, 256, 0, stream>>>(xw, batch, pooled, Nn);
    k_count<<<(Nn + 255) / 256, 256, 0, stream>>>(batch, cnt, Nn);
    k_div<<<(N_GRAPHS * HDIM + 255) / 256, 256, 0, stream>>>(pooled, cnt, N_GRAPHS * HDIM);

    // ---- MLP head ----
    k_mlp<<<(N_GRAPHS * 128 + 255) / 256, 256, 0, stream>>>(pooled, l1w, l1b, m1, N_GRAPHS, HDIM, 128);
    k_mlp<<<(N_GRAPHS * 3 + 255) / 256, 256, 0, stream>>>(m1, l2w, l2b, (float*)d_out, N_GRAPHS, 128, 3);
}
